// RAGATLayer_2276332667068
// MI455X (gfx1250) — matrix operand order repaired and checked by reading, not compiled, not run on hardware
//
#include <hip/hip_runtime.h>
#include <hip/hip_bf16.h>

#define N_NODES   50000
#define NUM_RELS  200
#define HEAD      2
#define DIM       128
#define E_REAL    400000
#define E_TOTAL   450000
#define NT_REAL   (E_REAL / 16)     // 25000 tiles of 16 real edges
#define NT_TOTAL  (E_TOTAL / 16)    // 28125 tiles total
#define GRID_EDGE 1024
#define REAL_BLKS 896               // blocks 0..895 -> real tiles, rest -> loop tiles
#define BN_EPS    1e-5f

typedef __attribute__((ext_vector_type(16))) __bf16        v16bf;
typedef __attribute__((ext_vector_type(8)))  float         v8f;
typedef __attribute__((ext_vector_type(4)))  float         v4f;
typedef __attribute__((ext_vector_type(4)))  unsigned int  v4u;

struct U32x8 { v4u lo, hi; };

// Two f32 -> packed bf16 pair (low = a, high = b), RNE, single VOP3 op.
static __device__ inline unsigned int pack2bf(float a, float b) {
  unsigned int r;
  asm("v_cvt_pk_bf16_f32 %0, %1, %2" : "=v"(r) : "v"(a), "v"(b));
  return r;
}

// LDS byte offset of a generic (__shared__) pointer: the LDS aperture maps
// addr[31:0] to the wave-relative LDS offset (ISA 10.2), and the async ops add
// LDS_BASE themselves, so truncation gives the VDST operand value.
static __device__ inline unsigned lds_off(const void* p) {
  return (unsigned)(unsigned long long)(uintptr_t)p;
}

// ---------------------------------------------------------------------------
// Edge message + attention + scatter kernel.
// Block = 64 threads (2 waves); wave h computes head h for the same 16-edge tile.
// msg[16x128] = x[16x128] @ W[128x128] via v_wmma_f32_16x16x32_bf16.
// W pre-staged once per block in LDS in B-fragment layout; the 16 N_embed rows
// of the NEXT tile are prefetched into a double-buffered LDS tile with
// global_load_async_to_lds_b128 (ASYNCcnt), overlapping the DMA with WMMA.
// The x tile is packed to bf16 in LDS (A layout) so the WMMA inner loop is
// pure ds_load_b128 + v_wmma with low register pressure (<256 VGPRs).
// ---------------------------------------------------------------------------
__global__ __launch_bounds__(64) void ragat_edge_kernel(
    const int* __restrict__ eidx,
    const float* __restrict__ nemb, const float* __restrict__ remb,
    const float* __restrict__ lemb, const float* __restrict__ wrel,
    const float* __restrict__ w1w,  const float* __restrict__ w1b,
    const float* __restrict__ w0w,  const float* __restrict__ w0b,
    const float* __restrict__ wtw,  const float* __restrict__ wtb,
    float* __restrict__ att_sum, float* __restrict__ agg)
{
  __shared__ unsigned int wfrag[16384];   // 64KB: [h][c][t][lane][8 u32] B fragments (bf16 pairs)
  __shared__ float nbuf[2][16][DIM];      // 16KB: double-buffered raw N_embed rows of a tile
  __shared__ unsigned int xbuf[2][1024];  // 8KB: per-head x tile [16 edges][64 u32] (bf16 pairs)
  __shared__ int   sdst[16], srel[16];
  __shared__ float attp[2][16];

  const int  tid  = threadIdx.x;
  const int  lane = tid & 31;
  const int  h    = tid >> 5;                       // wave id == head id
  const bool isReal = (blockIdx.x < REAL_BLKS);
  const float* __restrict__ Ww = isReal ? w1w : w0w;
  const float* __restrict__ Wb = isReal ? w1b : w0b;

  // --- Stage W (both heads) into LDS in WMMA B-fragment layout, bf16 ---
  // B (32x16 per wmma): lanes 0-15 hold K=0..15, lanes 16-31 hold K=16..31 of the
  // chunk; u32 j of a lane holds the bf16 pair {K=2j, K=2j+1}; N = 16*t + (lane&15).
  for (int idx = tid; idx < 16384; idx += 64) {
    int j  = idx & 7;
    int L  = (idx >> 3) & 31;
    int t  = (idx >> 8) & 7;
    int c  = (idx >> 11) & 3;
    int hh = (idx >> 13) & 1;
    int n  = 16 * t + (L & 15);
    int k0 = 32 * c + ((L < 16) ? 0 : 16) + 2 * j;
    float v0 = Ww[((size_t)hh * DIM + k0) * DIM + n];
    float v1 = Ww[((size_t)hh * DIM + k0 + 1) * DIM + n];
    wfrag[idx] = pack2bf(v0, v1);
  }
  // Per-lane constants: attention weights and output bias for this lane's 8 columns.
  float wt_t[8], wb_t[8];
#pragma unroll
  for (int t = 0; t < 8; ++t) {
    int n = 16 * t + (lane & 15);
    wt_t[t] = wtw[h * DIM + n];
    wb_t[t] = Wb[h * DIM + n];
  }
  const float wtbias = wtb[h];

  int tile0, tstride, tend;
  if (isReal) { tile0 = blockIdx.x;                         tstride = REAL_BLKS;             tend = NT_REAL;  }
  else        { tile0 = NT_REAL + (blockIdx.x - REAL_BLKS); tstride = GRID_EDGE - REAL_BLKS; tend = NT_TOTAL; }

  // Async-prefetch assignment: thread covers row tid>>2, 8 x 16B chunks.
  const int prow = tid >> 2;
  const int pchk = (tid & 3) * 8;      // first 16B-chunk index (of 32 per row)

  // --- prologue: async-load tile0's 16 N_embed rows into nbuf[0] ---
  {
    int srcn = eidx[2 * E_TOTAL + tile0 * 16 + prow];
    const float* gp = nemb + (size_t)srcn * DIM + pchk * 4;
    unsigned l0 = lds_off(&nbuf[0][prow][pchk * 4]);
#pragma unroll
    for (int i = 0; i < 8; ++i) {
      unsigned long long ga = (unsigned long long)(uintptr_t)(gp + i * 4);
      unsigned lo = l0 + i * 16;
      asm volatile("global_load_async_to_lds_b128 %0, %1, off"
                   :: "v"(lo), "v"(ga) : "memory");
    }
  }

  int buf = 0;
  for (int tile = tile0; tile < tend; tile += tstride) {
    // --- stage per-tile meta; wait for this tile's async N_embed rows ---
    if (tid < 16) {
      sdst[tid] = eidx[tile * 16 + tid];
      srel[tid] = eidx[E_TOTAL + tile * 16 + tid];
    }
    asm volatile("s_wait_asynccnt 0" ::: "memory");
    __syncthreads();

    // --- async-prefetch next tile's rows into the other buffer ---
    const int nxt = tile + tstride;
    if (nxt < tend) {
      int srcn = eidx[2 * E_TOTAL + nxt * 16 + prow];
      const float* gp = nemb + (size_t)srcn * DIM + pchk * 4;
      unsigned l0 = lds_off(&nbuf[buf ^ 1][prow][pchk * 4]);
#pragma unroll
      for (int i = 0; i < 8; ++i) {
        unsigned long long ga = (unsigned long long)(uintptr_t)(gp + i * 4);
        unsigned lo = l0 + i * 16;
        asm volatile("global_load_async_to_lds_b128 %0, %1, off"
                     :: "v"(lo), "v"(ga) : "memory");
      }
    }

    // --- Build x tile (bf16, A layout) for my head from the LDS-resident rows:
    //     x = Wrel[h,rel] * N_embed[src] * (R|L + 1). Lane owns edge m=lane&15,
    //     k-half kbase..kbase+63.
    {
      const int m     = lane & 15;
      const int kbase = (lane < 16) ? 0 : 64;
      const int rl    = srel[m];
      const float* np = &nbuf[buf][m][kbase];
      const float* wp = wrel + ((size_t)h * (NUM_RELS + 1) + rl) * DIM + kbase;
      const float* rp = (isReal ? (remb + (size_t)rl * DIM) : lemb) + kbase;
      unsigned int* xrow = &xbuf[h][m * 64 + (kbase >> 1)];
#pragma unroll
      for (int k = 0; k < 64; k += 4) {
        v4f nv = *(const v4f*)(np + k);
        v4f wv = *(const v4f*)(wp + k);
        v4f rv = *(const v4f*)(rp + k);
        v4f x  = wv * nv * (rv + 1.0f);
        xrow[(k >> 1)]     = pack2bf(x.x, x.y);
        xrow[(k >> 1) + 1] = pack2bf(x.z, x.w);
      }
    }
    __syncthreads();

    // --- WMMA: acc[t] (16x16 f32) accumulates over 4 K-chunks of 32 ---
    v8f acc[8];
#pragma unroll
    for (int t = 0; t < 8; ++t)
#pragma unroll
      for (int r = 0; r < 8; ++r) acc[t][r] = 0.0f;

    const unsigned int* xb = &xbuf[h][0];
    const int am = lane & 15;
#pragma unroll
    for (int c = 0; c < 4; ++c) {
      // A (16x32): lane<16 row=lane holds K=0..15, lane>=16 K=16..31
      const int kb = 32 * c + ((lane < 16) ? 0 : 16);
      U32x8 ar;
      ar.lo = *(const v4u*)(xb + am * 64 + (kb >> 1));
      ar.hi = *(const v4u*)(xb + am * 64 + (kb >> 1) + 4);
      v16bf a = __builtin_bit_cast(v16bf, ar);
#pragma unroll
      for (int t = 0; t < 8; ++t) {
        const unsigned int* bp = &wfrag[(((h * 4 + c) * 8 + t) * 32 + lane) * 8];
        U32x8 br; br.lo = *(const v4u*)(bp); br.hi = *(const v4u*)(bp + 4);
        v16bf b = __builtin_bit_cast(v16bf, br);
        acc[t] = __builtin_amdgcn_wmma_f32_16x16x32_bf16(
            false, a, false, b, (short)0, acc[t], false, false);
      }
    }

    // --- add bias (per output column n) ---
#pragma unroll
    for (int t = 0; t < 8; ++t)
#pragma unroll
      for (int r = 0; r < 8; ++r) acc[t][r] += wb_t[t];

    // --- attention logit per edge row: sum_n msg[m][n]*Wt[n] via LDS f32 atomics ---
    if (lane < 16) attp[h][lane] = 0.0f;
    __syncthreads();
#pragma unroll
    for (int r = 0; r < 8; ++r) {
      float p = 0.0f;
#pragma unroll
      for (int t = 0; t < 8; ++t) p += acc[t][r] * wt_t[t];
      int mm = (lane < 16) ? r : (r + 8);
      atomicAdd(&attp[h][mm], p);
    }
    __syncthreads();
    if (lane < 16) {
      float logit = attp[h][lane] + wtbias;
      float lr = (logit > 0.0f) ? logit : 0.2f * logit;
      attp[h][lane] = __expf(-lr);
    }
    __syncthreads();

    // --- scatter: agg[h,dst] += att*msg ; att_sum[h,dst] += att ---
#pragma unroll
    for (int r = 0; r < 8; ++r) {
      int   mm = (lane < 16) ? r : (r + 8);
      float a  = attp[h][mm];
      int   node = sdst[mm];
      float* aggrow = agg + ((size_t)(h * N_NODES + node)) * DIM;
      if ((lane & 15) == 0) atomicAdd(&att_sum[h * N_NODES + node], a);
#pragma unroll
      for (int t = 0; t < 8; ++t) {
        int n = 16 * t + (lane & 15);
        atomicAdd(aggrow + n, a * acc[t][r]);
      }
    }
    __syncthreads();
    buf ^= 1;
  }
}

// ---------------------------------------------------------------------------
__global__ void zero_kernel(float* __restrict__ p, long long n) {
  long long i = (long long)blockIdx.x * blockDim.x + threadIdx.x;
  long long s = (long long)gridDim.x * blockDim.x;
  for (; i < n; i += s) p[i] = 0.0f;
}

// h = mean_h(agg/att_sum) + bias; accumulate BN column sums (two-level reduction)
__global__ __launch_bounds__(128) void node_bn_stats(
    const float* __restrict__ att_sum, const float* __restrict__ agg,
    const float* __restrict__ bias, float* __restrict__ hbuf,
    float* __restrict__ musum, float* __restrict__ sqsum)
{
  const int n = threadIdx.x;  // column 0..127
  float lsum = 0.0f, lsq = 0.0f;
  const float bn = bias[n];
  for (int i = blockIdx.x; i < N_NODES; i += gridDim.x) {
    float s0 = att_sum[i];
    float s1 = att_sum[N_NODES + i];
    s0 = (s0 == 0.0f) ? 1.0f : s0;
    s1 = (s1 == 0.0f) ? 1.0f : s1;
    float a0 = agg[(size_t)i * DIM + n];
    float a1 = agg[((size_t)N_NODES + i) * DIM + n];
    float hv = 0.5f * (a0 / s0 + a1 / s1) + bn;
    hbuf[(size_t)i * DIM + n] = hv;
    lsum += hv;
    lsq  += hv * hv;
  }
  atomicAdd(&musum[n], lsum);
  atomicAdd(&sqsum[n], lsq);
}

__global__ void bn_tanh_kernel(
    const float* __restrict__ hbuf, const float* __restrict__ musum,
    const float* __restrict__ sqsum, const float* __restrict__ gamma,
    const float* __restrict__ beta, float* __restrict__ nout)
{
  const long long total = (long long)N_NODES * DIM;
  long long i = (long long)blockIdx.x * blockDim.x + threadIdx.x;
  long long s = (long long)gridDim.x * blockDim.x;
  const float invN = 1.0f / (float)N_NODES;
  for (; i < total; i += s) {
    int n = (int)(i & (DIM - 1));
    float mu  = musum[n] * invN;
    float var = sqsum[n] * invN - mu * mu;
    float sc  = gamma[n] * rsqrtf(var + BN_EPS);
    float v   = (hbuf[i] - mu) * sc + beta[n];
    nout[i] = tanhf(v);
  }
}

__global__ void rout_kernel(
    const float* __restrict__ remb, const float* __restrict__ wrw,
    const float* __restrict__ wrb, float* __restrict__ rout)
{
  int idx = blockIdx.x * blockDim.x + threadIdx.x;
  if (idx >= NUM_RELS * DIM) return;
  int r = idx / DIM, o = idx % DIM;
  float acc = wrb[o];
  const float* a = remb + (size_t)r * DIM;
  const float* w = wrw + (size_t)o * DIM;
  for (int i = 0; i < DIM; ++i) acc += a[i] * w[i];
  rout[idx] = acc;
}

// ---------------------------------------------------------------------------
extern "C" void kernel_launch(void* const* d_in, const int* in_sizes, int n_in,
                              void* d_out, int out_size, void* d_ws, size_t ws_size,
                              hipStream_t stream) {
  const int*   eidx  = (const int*)d_in[0];
  // d_in[1] = num_edges (known statically: E_REAL)
  const float* nemb  = (const float*)d_in[2];
  const float* remb  = (const float*)d_in[3];
  const float* lemb  = (const float*)d_in[4];
  const float* wrel  = (const float*)d_in[5];
  const float* w1w   = (const float*)d_in[6];
  const float* w1b   = (const float*)d_in[7];
  const float* w0w   = (const float*)d_in[8];
  const float* w0b   = (const float*)d_in[9];
  const float* wtw   = (const float*)d_in[10];
  const float* wtb   = (const float*)d_in[11];
  const float* bias  = (const float*)d_in[12];
  const float* gamma = (const float*)d_in[13];
  const float* beta  = (const float*)d_in[14];
  const float* wrw   = (const float*)d_in[15];
  const float* wrb   = (const float*)d_in[16];

  float* ws      = (float*)d_ws;
  float* att_sum = ws;                                   // 2*50000
  float* agg     = att_sum + 2 * N_NODES;                // 2*50000*128
  float* hbuf    = agg + (size_t)2 * N_NODES * DIM;      // 50000*128
  float* musum   = hbuf + (size_t)N_NODES * DIM;         // 128
  float* sqsum   = musum + DIM;                          // 128
  float* nout    = (float*)d_out;
  float* rout    = nout + (size_t)N_NODES * DIM;

  const long long nzero = 2LL * N_NODES + 2LL * N_NODES * DIM;
  zero_kernel<<<2048, 256, 0, stream>>>(ws, nzero);
  zero_kernel<<<1, 256, 0, stream>>>(musum, 2LL * DIM);

  ragat_edge_kernel<<<GRID_EDGE, 64, 0, stream>>>(
      eidx, nemb, remb, lemb, wrel, w1w, w1b, w0w, w0b, wtw, wtb, att_sum, agg);

  node_bn_stats<<<512, 128, 0, stream>>>(att_sum, agg, bias, hbuf, musum, sqsum);
  bn_tanh_kernel<<<8192, 256, 0, stream>>>(hbuf, musum, sqsum, gamma, beta, nout);
  rout_kernel<<<(NUM_RELS * DIM + 255) / 256, 256, 0, stream>>>(remb, wrw, wrb, rout);
}
